// CrossAttention_42047729828228
// MI455X (gfx1250) — compile-verified, matrix operand order checked by reading, not run on hardware
//
#include <hip/hip_runtime.h>
#include <hip/hip_bf16.h>
#include <math.h>

// ---------------------------------------------------------------------------
// CDNA5 (gfx1250) fused cross-attention pipeline, all GEMMs on
// v_wmma_f32_16x16x32_f16 (wave32), async global->LDS staging where available.
//
// Shapes: B=32, N=1536, M=80 (padded to 96 for the PV gemm), H=8, C=64,
// QD=320, KD=VD=768, inner=512.  All K-dims are multiples of 32.
// ---------------------------------------------------------------------------

typedef __attribute__((ext_vector_type(16))) _Float16 v16h;
typedef __attribute__((ext_vector_type(8)))  float    v8f;
typedef __attribute__((ext_vector_type(4)))  int      v4i;

union HFrag { v16h v; unsigned u[8]; };

__device__ inline unsigned short f2h_bits(float f) {
  union { _Float16 h; unsigned short u; } cv;
  cv.h = (_Float16)f;
  return cv.u;
}

__device__ inline v8f zero8() {
  v8f z;
#pragma unroll
  for (int i = 0; i < 8; ++i) z[i] = 0.0f;
  return z;
}

// 16-byte global -> LDS copy; async path on gfx1250 (GLOBAL_LOAD_ASYNC_TO_LDS,
// tracked by ASYNCcnt), plain vector copy otherwise.
__device__ inline void copy16_g2lds(void* lds_dst, const void* gsrc) {
#if __has_builtin(__builtin_amdgcn_global_load_async_to_lds_b128)
  __builtin_amdgcn_global_load_async_to_lds_b128(
      (__attribute__((address_space(1))) v4i*)(gsrc),
      (__attribute__((address_space(3))) v4i*)(lds_dst), 0, 0);
#else
  *(uint4*)lds_dst = *(const uint4*)gsrc;
#endif
}

__device__ inline void copy16_wait() {
#if __has_builtin(__builtin_amdgcn_s_wait_asynccnt)
  __builtin_amdgcn_s_wait_asynccnt(0);
#endif
}

// Load a 16x32 f16 A/B fragment from an LDS tile.
// ISA 7.12.2 layout: lane L (0..15) holds row L, K = {0..7,16..23};
// lane L+16 holds row L, K = {8..15, 24..31}.  Each pair of halves is one u32.
__device__ inline v16h load_frag16(const unsigned short* base, int rowBase,
                                   int rowStride, int lane) {
  HFrag f;
  const int r  = rowBase + (lane & 15);
  const int kb = (lane >> 4) << 3;                 // 0 or 8
  const unsigned short* p = base + r * rowStride + kb;
#pragma unroll
  for (int g = 0; g < 8; ++g) {
    const int k0 = ((g & 3) << 1) + ((g >> 2) << 4);  // 0,2,4,6,16,18,20,22
    f.u[g] = *(const unsigned*)(p + k0);
  }
  return f.v;
}

__global__ void zero_u16_kernel(unsigned short* p, int n) {
  int i = blockIdx.x * blockDim.x + threadIdx.x;
  if (i < n) p[i] = 0;
}

// ---------------------------------------------------------------------------
// Generic C = A * B^T GEMM.  A: (Ma x K) row-major (f32 if AHALF==0, f16 if
// AHALF==1), B: (Nb x K) f32 row-major weights.  64x64 tile per 128-thread
// block (4 waves, each wave owns a 16x64 strip = 4 WMMA accumulators).
// CMODE 0: C f32 row-major + bias
// CMODE 1: C f16 row-major
// CMODE 2: C f16 scattered as per-head transposed V: (b,h,c,m) -> [..]*96+m
// ---------------------------------------------------------------------------
template <int AHALF, int CMODE>
__global__ void __launch_bounds__(128) gemm_bt_kernel(
    const void* __restrict__ A, int lda, const float* __restrict__ Bw, int ldb,
    void* __restrict__ C, int ldc, const float* __restrict__ bias, int K) {
  __shared__ __align__(16) unsigned short As[64 * 32];
  __shared__ __align__(16) unsigned short Bs[64 * 32];

  const int tid  = threadIdx.x;
  const int wave = tid >> 5;
  const int lane = tid & 31;
  const int m0   = blockIdx.y << 6;
  const int n0   = blockIdx.x << 6;

  v8f acc[4];
#pragma unroll
  for (int j = 0; j < 4; ++j) acc[j] = zero8();

  for (int k0 = 0; k0 < K; k0 += 32) {
    __syncthreads();
    if (AHALF) {
      const unsigned short* Ah = (const unsigned short*)A;
#pragma unroll
      for (int cidx = tid; cidx < 256; cidx += 128) {  // 16B chunks (8 halves)
        const int row = cidx >> 2, col = (cidx & 3) << 3;
        *(uint4*)&As[(row << 5) + col] =
            *(const uint4*)&Ah[(size_t)(m0 + row) * lda + k0 + col];
      }
    } else {
      const float* Af = (const float*)A;
#pragma unroll
      for (int cidx = tid; cidx < 512; cidx += 128) {  // 16B chunks (4 f32)
        const int row = cidx >> 3, col = (cidx & 7) << 2;
        const float4 f =
            *(const float4*)&Af[(size_t)(m0 + row) * lda + k0 + col];
        union { unsigned short s[4]; uint2 v; } pk;
        pk.s[0] = f2h_bits(f.x); pk.s[1] = f2h_bits(f.y);
        pk.s[2] = f2h_bits(f.z); pk.s[3] = f2h_bits(f.w);
        *(uint2*)&As[(row << 5) + col] = pk.v;
      }
    }
#pragma unroll
    for (int cidx = tid; cidx < 512; cidx += 128) {
      const int row = cidx >> 3, col = (cidx & 7) << 2;
      const float4 f =
          *(const float4*)&Bw[(size_t)(n0 + row) * ldb + k0 + col];
      union { unsigned short s[4]; uint2 v; } pk;
      pk.s[0] = f2h_bits(f.x); pk.s[1] = f2h_bits(f.y);
      pk.s[2] = f2h_bits(f.z); pk.s[3] = f2h_bits(f.w);
      *(uint2*)&Bs[(row << 5) + col] = pk.v;
    }
    __syncthreads();
    const v16h a = load_frag16(As, wave << 4, 32, lane);
#pragma unroll
    for (int j = 0; j < 4; ++j) {
      const v16h b = load_frag16(Bs, j << 4, 32, lane);
      acc[j] = __builtin_amdgcn_wmma_f32_16x16x32_f16(
          false, a, false, b, (short)0, acc[j], false, false);
    }
  }

  const int lhi = lane >> 4, lcol = lane & 15;
#pragma unroll
  for (int j = 0; j < 4; ++j) {
#pragma unroll
    for (int r = 0; r < 8; ++r) {
      const int row = m0 + (wave << 4) + r + (lhi << 3);
      const int col = n0 + (j << 4) + lcol;
      const float v = acc[j][r];
      if (CMODE == 0) {
        ((float*)C)[(size_t)row * ldc + col] = v + bias[col];
      } else if (CMODE == 1) {
        ((unsigned short*)C)[(size_t)row * ldc + col] = f2h_bits(v);
      } else {  // scatter: row = b*80+m over B*M, col = h*64+ch
        const int b = row / 80, m = row - b * 80;
        const int h = col >> 6, ch = col & 63;
        ((unsigned short*)C)[((size_t)(((b << 3) + h) << 6 | ch)) * 96 + m] =
            f2h_bits(v);
      }
    }
  }
}

// ---------------------------------------------------------------------------
// Fused attention: one block = one (b, h, 64-row n-tile).
// S = q k^T (two K=32 steps over C=64, 5 column tiles over M=80),
// epilogue: *0.125, mask-fill, +5*box +5*road, LDS softmax over M (pad to 96),
// O = P vT^T (three K=32 steps over Mpad=96, 4 column tiles over C=64).
// Writes O half into the q buffer (exclusive row/head-column region per block).
// ---------------------------------------------------------------------------
__global__ void __launch_bounds__(128) attn_kernel(
    const unsigned short* __restrict__ qh,   // (B*N, 512) half
    const unsigned short* __restrict__ kh,   // (B*M, 512) half
    const unsigned short* __restrict__ vT,   // (B,H,64,96) half, pad zeroed
    const unsigned char* __restrict__ mask,  // (B, M) bool
    const float* __restrict__ box,           // (B, N, M)
    const float* __restrict__ road,          // (B, 1, N)
    unsigned short* __restrict__ oh) {       // (B*N, 512) half (aliases qh)
  __shared__ __align__(16) unsigned short qs[64 * 64];
  __shared__ __align__(16) unsigned short kss[80 * 64];
  __shared__ __align__(16) unsigned short vs[64 * 96];
  __shared__ __align__(16) float          Ss[64 * 80];
  __shared__ __align__(16) unsigned short Ps[64 * 96];

  const int tid  = threadIdx.x;
  const int wave = tid >> 5;
  const int lane = tid & 31;
  const int b    = blockIdx.y >> 3;
  const int h    = blockIdx.y & 7;
  const int nb   = blockIdx.x << 6;
  const int N = 1536, M = 80;

  // Stage q (64x64), k (80x64), vT (first 32 of 64 rows, 32x96) halves into LDS, 16B granules.
#pragma unroll
  for (int i = tid; i < 64 * 8; i += 128) {
    const int r = i >> 3, c8 = (i & 7) << 3;
    copy16_g2lds(&qs[(r << 6) + c8],
                 &qh[(size_t)(b * N + nb + r) * 512 + (h << 6) + c8]);
  }
#pragma unroll
  for (int i = tid; i < 80 * 8; i += 128) {
    const int r = i >> 3, c8 = (i & 7) << 3;
    copy16_g2lds(&kss[(r << 6) + c8],
                 &kh[(size_t)(b * M + r) * 512 + (h << 6) + c8]);
  }
  {
    const unsigned short* src = vT + (size_t)((b << 3) + h) * 64 * 96;
#pragma unroll
    for (int i = tid; i < 64 * 6; i += 128) {
      copy16_g2lds(&vs[i << 3], &src[(size_t)i << 3]);
    }
  }
  copy16_wait();
  __syncthreads();

  // ---- S = q k^T --------------------------------------------------------
  v8f sacc[5];
#pragma unroll
  for (int t = 0; t < 5; ++t) sacc[t] = zero8();
#pragma unroll
  for (int kk = 0; kk < 2; ++kk) {
    const v16h a = load_frag16(qs + kk * 32, wave << 4, 64, lane);
#pragma unroll
    for (int t = 0; t < 5; ++t) {
      const v16h bf = load_frag16(kss + kk * 32, t << 4, 64, lane);
      sacc[t] = __builtin_amdgcn_wmma_f32_16x16x32_f16(
          false, a, false, bf, (short)0, sacc[t], false, false);
    }
  }

  // ---- scale, mask, additive biases -> Ss -------------------------------
  const int lhi = lane >> 4, lcol = lane & 15;
#pragma unroll
  for (int t = 0; t < 5; ++t) {
#pragma unroll
    for (int r = 0; r < 8; ++r) {
      const int rowl = (wave << 4) + r + (lhi << 3);  // 0..63
      const int m    = (t << 4) + lcol;               // 0..79
      const int n    = nb + rowl;
      float s = sacc[t][r] * 0.125f;                  // DIM_HEAD^-0.5
      if (!mask[b * M + m]) s = -3.4028234663852886e+38f;
      s += 5.0f * box[((size_t)b * N + n) * M + m] +
           5.0f * road[(size_t)b * N + n];
      Ss[rowl * 80 + m] = s;
    }
  }
  __syncthreads();

  // ---- row softmax over M, write P (half, padded to 96) -----------------
  if (tid < 64) {
    const int rowl = tid;
    float mx = -3.4028234663852886e+38f;
    for (int m = 0; m < 80; ++m) mx = fmaxf(mx, Ss[rowl * 80 + m]);
    float sum = 0.0f;
    for (int m = 0; m < 80; ++m) sum += __expf(Ss[rowl * 80 + m] - mx);
    const float inv = 1.0f / sum;
    for (int m = 0; m < 80; ++m)
      Ps[rowl * 96 + m] = f2h_bits(__expf(Ss[rowl * 80 + m] - mx) * inv);
    for (int m = 80; m < 96; ++m) Ps[rowl * 96 + m] = 0;
  }
  __syncthreads();

  // ---- O = P vT^T -------------------------------------------------------
  v8f oacc[4];
#pragma unroll
  for (int j = 0; j < 4; ++j) oacc[j] = zero8();
#pragma unroll
  for (int kk = 0; kk < 3; ++kk) {
    const v16h a = load_frag16(Ps + kk * 32, wave << 4, 96, lane);
#pragma unroll
    for (int j = 0; j < 4; ++j) {
      const v16h bf = load_frag16(vs + kk * 32, j << 4, 96, lane);
      oacc[j] = __builtin_amdgcn_wmma_f32_16x16x32_f16(
          false, a, false, bf, (short)0, oacc[j], false, false);
    }
  }
#pragma unroll
  for (int j = 0; j < 4; ++j) {
#pragma unroll
    for (int r = 0; r < 8; ++r) {
      const int rowl = (wave << 4) + r + (lhi << 3);
      const int c    = (j << 4) + lcol;
      oh[(size_t)(b * N + nb + rowl) * 512 + (h << 6) + c] =
          f2h_bits(oacc[j][r]);
    }
  }
}

// ---------------------------------------------------------------------------
extern "C" void kernel_launch(void* const* d_in, const int* in_sizes, int n_in,
                              void* d_out, int out_size, void* d_ws,
                              size_t ws_size, hipStream_t stream) {
  const float*         x     = (const float*)d_in[0];
  const float*         key   = (const float*)d_in[1];
  const float*         value = (const float*)d_in[2];
  const unsigned char* mask  = (const unsigned char*)d_in[3];
  const float*         box   = (const float*)d_in[4];
  const float*         road  = (const float*)d_in[5];
  const float*         Wq    = (const float*)d_in[6];
  const float*         Wk    = (const float*)d_in[7];
  const float*         Wv    = (const float*)d_in[8];
  const float*         Wo    = (const float*)d_in[9];
  const float*         bo    = (const float*)d_in[10];
  float*               out   = (float*)d_out;

  // workspace layout (half): q/o aliased (50.3 MB) | k (2.6 MB) | vT (3.1 MB)
  unsigned short* qh = (unsigned short*)d_ws;
  unsigned short* kh = (unsigned short*)((char*)d_ws + (size_t)49152 * 512 * 2);
  unsigned short* vT = (unsigned short*)((char*)d_ws + (size_t)49152 * 512 * 2 +
                                         (size_t)2560 * 512 * 2);

  const dim3 blk(128);

  // 0) zero vT (needed for the M=80 -> 96 pad columns)
  const int nvT = 32 * 8 * 64 * 96;
  zero_u16_kernel<<<(nvT + 255) / 256, 256, 0, stream>>>(vT, nvT);

  // 1) q = x @ Wq^T  -> half (49152 x 512), K=320
  gemm_bt_kernel<0, 1><<<dim3(8, 768), blk, 0, stream>>>(
      x, 320, Wq, 320, qh, 512, nullptr, 320);

  // 2) k = key @ Wk^T -> half (2560 x 512), K=768
  gemm_bt_kernel<0, 1><<<dim3(8, 40), blk, 0, stream>>>(
      key, 768, Wk, 768, kh, 512, nullptr, 768);

  // 3) v = value @ Wv^T -> per-head transposed half (B,H,64,96), K=768
  gemm_bt_kernel<0, 2><<<dim3(8, 40), blk, 0, stream>>>(
      value, 768, Wv, 768, vT, 0, nullptr, 768);

  // 4) fused attention: (n-tiles=24) x (B*H=256)
  attn_kernel<<<dim3(24, 256), blk, 0, stream>>>(qh, kh, vT, mask, box, road,
                                                 qh);

  // 5) out = o @ Wo^T + bo -> f32 (49152 x 320), K=512
  gemm_bt_kernel<1, 0><<<dim3(5, 768), blk, 0, stream>>>(
      qh, 512, Wo, 512, out, 320, bo, 512);
}
